// SineLayer_68805376082638
// MI455X (gfx1250) — matrix operand order repaired and checked by reading, not compiled, not run on hardware
//
#include <hip/hip_runtime.h>
#include <hip/hip_bf16.h>

typedef __attribute__((ext_vector_type(16))) _Float16 v16h;
typedef __attribute__((ext_vector_type(8)))  _Float16 v8h;
typedef __attribute__((ext_vector_type(8)))  float    v8f;

#define BATCH 32768
#define NIN   256
#define NOUT  512
#define DD    16
#define KTOT  (NIN*DD)   /* 4096 */
#define BM    128
#define BN    128
#define BK    32         /* one WMMA K step = 2 input features * 16 basis fns */
#define NITER (KTOT/BK)  /* 128 */
#define LDP   (BK+8)     /* padded LDS row: 40 halves = 80B */
#define EPSV  1e-8f

__global__ __launch_bounds__(256) void sine_layer_wmma(
    const float* __restrict__ X,      // (BATCH, NIN)
    const float* __restrict__ omega,  // (DD)
    const float* __restrict__ phase,  // (DD)
    const float* __restrict__ C,      // (NOUT, KTOT) row-major
    const float* __restrict__ bias,   // (NOUT)
    float* __restrict__ Y)            // (BATCH, NOUT)
{
    __shared__ __align__(16) _Float16 Alds[2][BM][LDP];
    __shared__ __align__(16) _Float16 Blds[2][BN][LDP];
    __shared__ float4 cS[DD];   // per-d: {w, p, mu*inv, inv}

    const int tid   = threadIdx.x;
    const int mbase = blockIdx.x * BM;
    const int nbase = blockIdx.y * BN;

    // ---- one-time basis statistics: mu = e^{-w^2/2} sin p ; std = sqrt(.5(1-e^{-2w^2}cos 2p)-mu^2)
    if (tid < DD) {
        float w  = omega[tid];
        float p  = phase[tid];
        float mu = __expf(-0.5f * w * w) * __sinf(p);
        float vr = 0.5f * (1.0f - __expf(-2.0f * w * w) * __cosf(2.0f * p)) - mu * mu;
        float iv = 1.0f / (sqrtf(fmaxf(vr, 0.0f)) + EPSV);
        cS[tid]  = make_float4(w, p, mu * iv, iv);
    }

    // staging roles: thread t handles (row = t&127, seg = t>>7) for both A and B tiles
    const int srow = tid & (BM - 1);
    const int sseg = tid >> 7;            // 0..1 -> k sub-range seg*16

    // wave tiling: 8 waves in 4(M) x 2(N)
    const int lane = tid & 31;
    const int wv   = tid >> 5;
    const int wm   = wv & 3;              // 0..3 -> 32-row strip
    const int wn   = wv >> 2;             // 0..1 -> 64-col strip
    const int l16  = lane & 15;
    const int hif  = lane >> 4;           // 0 or 1 (upper half-wave)

    v8f acc[2][4];
    #pragma unroll
    for (int mi = 0; mi < 2; ++mi)
        #pragma unroll
        for (int ni = 0; ni < 4; ++ni)
            acc[mi][ni] = (v8f){0.f,0.f,0.f,0.f,0.f,0.f,0.f,0.f};

    __syncthreads();   // cS visible

    const size_t xrow_off = (size_t)(mbase + srow) * NIN;
    const size_t crow_off = (size_t)(nbase + srow) * KTOT + sseg * DD;

    // stage tile at k-offset kk into registers (global loads + basis eval + f16 convert)
    auto stage = [&](int kk, v8h& ah0, v8h& ah1, v8h& bh0, v8h& bh1) {
        const float xv = X[xrow_off + (kk / DD) + sseg];
        #pragma unroll
        for (int d = 0; d < 8; ++d) {
            float4 cc = cS[d];
            ah0[d] = (_Float16)fmaf(__sinf(fmaf(cc.x, xv, cc.y)), cc.w, cc.z);
        }
        #pragma unroll
        for (int d = 8; d < DD; ++d) {
            float4 cc = cS[d];
            ah1[d - 8] = (_Float16)fmaf(__sinf(fmaf(cc.x, xv, cc.y)), cc.w, cc.z);
        }
        const float4* cg = (const float4*)(C + crow_off + kk);
        float fb[16];
        *(float4*)&fb[0]  = cg[0];
        *(float4*)&fb[4]  = cg[1];
        *(float4*)&fb[8]  = cg[2];
        *(float4*)&fb[12] = cg[3];
        if (kk + BK < KTOT) __builtin_prefetch(C + crow_off + kk + BK, 0, 3);
        #pragma unroll
        for (int j = 0; j < 8; ++j) { bh0[j] = (_Float16)fb[j]; bh1[j] = (_Float16)fb[j + 8]; }
    };

    // ---- prologue: stage tile 0 into LDS buffer 0
    {
        v8h ah0, ah1, bh0, bh1;
        stage(0, ah0, ah1, bh0, bh1);
        *(v8h*)&Alds[0][srow][sseg * DD]     = ah0;
        *(v8h*)&Alds[0][srow][sseg * DD + 8] = ah1;
        *(v8h*)&Blds[0][srow][sseg * DD]     = bh0;
        *(v8h*)&Blds[0][srow][sseg * DD + 8] = bh1;
    }
    __syncthreads();

    // ---- main loop: 1 barrier per iteration (double-buffered LDS)
    #pragma unroll 2
    for (int it = 0; it < NITER; ++it) {
        const int cur = it & 1;
        const int nxt = cur ^ 1;

        // fragments for this tile (contiguous k = 16*half + s, same for A and B), all in distinct registers
        v16h af[2];
        #pragma unroll
        for (int mi = 0; mi < 2; ++mi) {
            const int row = wm * 32 + mi * 16 + l16;
            const int k0  = hif * 16;                     // lanes>=16: K=16..31
            v8h a0 = *(const v8h*)&Alds[cur][row][k0];
            v8h a1 = *(const v8h*)&Alds[cur][row][k0 + 8];
            #pragma unroll
            for (int j = 0; j < 8; ++j) { af[mi][j] = a0[j]; af[mi][j + 8] = a1[j]; }
        }
        v16h bf[4];
        #pragma unroll
        for (int ni = 0; ni < 4; ++ni) {
            const int col = wn * 64 + ni * 16 + l16;
            const int k0  = hif * 16;                     // lanes>=16: K=16..31
            v8h b0 = *(const v8h*)&Blds[cur][col][k0];
            v8h b1 = *(const v8h*)&Blds[cur][col][k0 + 8];
            #pragma unroll
            for (int j = 0; j < 8; ++j) { bf[ni][j] = b0[j]; bf[ni][j + 8] = b1[j]; }
        }

        // stage next tile into the other buffer (overlaps with fragment-load latency;
        // writes target the buffer whose readers all synced at the previous barrier)
        if (it + 1 < NITER) {
            v8h ah0, ah1, bh0, bh1;
            stage((it + 1) * BK, ah0, ah1, bh0, bh1);
            *(v8h*)&Alds[nxt][srow][sseg * DD]     = ah0;
            *(v8h*)&Alds[nxt][srow][sseg * DD + 8] = ah1;
            *(v8h*)&Blds[nxt][srow][sseg * DD]     = bh0;
            *(v8h*)&Blds[nxt][srow][sseg * DD + 8] = bh1;
        }

        // 8 back-to-back WMMAs, no intra-burst LDS dependencies
        #pragma unroll
        for (int ni = 0; ni < 4; ++ni)
            #pragma unroll
            for (int mi = 0; mi < 2; ++mi)
                acc[mi][ni] = __builtin_amdgcn_wmma_f32_16x16x32_f16(
                    false, af[mi], false, bf[ni], (short)0, acc[mi][ni], false, false);

        __syncthreads();
    }

    // ---------- epilogue: D layout -> global, add bias
    #pragma unroll
    for (int ni = 0; ni < 4; ++ni) {
        const int col = nbase + wn * 64 + ni * 16 + l16;
        const float bv = bias[col];
        #pragma unroll
        for (int mi = 0; mi < 2; ++mi) {
            const int rbase = mbase + wm * 32 + mi * 16 + hif * 8;
            #pragma unroll
            for (int r = 0; r < 8; ++r) {
                Y[(size_t)(rbase + r) * NOUT + col] = acc[mi][ni][r] + bv;
            }
        }
    }
}

extern "C" void kernel_launch(void* const* d_in, const int* in_sizes, int n_in,
                              void* d_out, int out_size, void* d_ws, size_t ws_size,
                              hipStream_t stream) {
    const float* X     = (const float*)d_in[0];
    const float* omega = (const float*)d_in[1];
    const float* phase = (const float*)d_in[2];
    const float* C     = (const float*)d_in[3];
    const float* bias  = (const float*)d_in[4];
    float* Y = (float*)d_out;

    dim3 grid(BATCH / BM, NOUT / BN);   // 256 x 4
    sine_layer_wmma<<<grid, dim3(256), 0, stream>>>(X, omega, phase, C, bias, Y);
}
